// SAModule_61847529062859
// MI455X (gfx1250) — matrix operand order repaired and checked by reading, not compiled, not run on hardware
//
#include <hip/hip_runtime.h>
#include <hip/hip_bf16.h>

typedef __attribute__((ext_vector_type(16))) _Float16 v16h;
typedef __attribute__((ext_vector_type(8)))  _Float16 v8h;
typedef __attribute__((ext_vector_type(8)))  float    v8f;
typedef __attribute__((ext_vector_type(4)))  float    v4f;

#define BCLOUD 4
#define MPTS   4096
#define NPTS   (BCLOUD * MPTS)
#define KNBR   64
#define FIN    64
#define F1     64
#define F2     64
#define F3     128
#define RADIUS2 0.04f      // R = 0.2

// padded LDS row strides (halfs): 16B-aligned, bank-stagger
#define W1S 104            // 96 + 8
#define W2S 72             // 64 + 8
#define W3S 72
#define HS  72             // H rows: 64 + 8

// ---------------------------------------------------------------------------
// Fragment loaders (CDNA5 ISA 7.12.2, wave32, 16-bit operands).
// A 16x32: lane m needs k = kc*32 + half*8 + {0..7}  and  +16 + {0..7}
//          -> two contiguous groups of 8 halfs (two ds_load_b128).
// B 32x16: lane n needs k = kc*32 + half*8 + {0..7}  and  +16 + {0..7}
//          -> two contiguous groups of 8 halfs (two ds_load_b128).
// ---------------------------------------------------------------------------
__device__ __forceinline__ v16h ldsA(const _Float16* p) {
  v8h lo = *(const v8h*)(p);        // k group 0
  v8h hi = *(const v8h*)(p + 16);   // k group 1
  return __builtin_shufflevector(lo, hi, 0, 1, 2, 3, 4, 5, 6, 7,
                                         8, 9, 10, 11, 12, 13, 14, 15);
}
__device__ __forceinline__ v16h ldsB(const _Float16* p) {
  v8h lo = *(const v8h*)(p);
  v8h hi = *(const v8h*)(p + 16);
  return __builtin_shufflevector(lo, hi, 0, 1, 2, 3, 4, 5, 6, 7,
                                         8, 9, 10, 11, 12, 13, 14, 15);
}
// A-fragment from f32 global: two groups of 8 floats -> f16
__device__ __forceinline__ v16h cvtA(v4f a0, v4f a1, v4f b0, v4f b1) {
  v16h r;
  r[0]  = (_Float16)a0[0]; r[1]  = (_Float16)a0[1];
  r[2]  = (_Float16)a0[2]; r[3]  = (_Float16)a0[3];
  r[4]  = (_Float16)a1[0]; r[5]  = (_Float16)a1[1];
  r[6]  = (_Float16)a1[2]; r[7]  = (_Float16)a1[3];
  r[8]  = (_Float16)b0[0]; r[9]  = (_Float16)b0[1];
  r[10] = (_Float16)b0[2]; r[11] = (_Float16)b0[3];
  r[12] = (_Float16)b1[0]; r[13] = (_Float16)b1[1];
  r[14] = (_Float16)b1[2]; r[15] = (_Float16)b1[3];
  return r;
}

// ---------------------------------------------------------------------------
// Kernel 1: radius neighbor search, first-K-in-ball (deterministic).
// ---------------------------------------------------------------------------
__global__ __launch_bounds__(256) void nbr_kernel(const float* __restrict__ pos,
                                                  int* __restrict__ nbr,
                                                  int* __restrict__ cnt) {
  int q = blockIdx.x * 256 + threadIdx.x;
  if (q >= NPTS) return;
  int base = q & ~(MPTS - 1);
  float qx = pos[q * 3 + 0];
  float qy = pos[q * 3 + 1];
  float qz = pos[q * 3 + 2];
  int c = 0;
  for (int j = 0; j < MPTS; ++j) {
    int p = base + j;
    float dx = pos[p * 3 + 0] - qx;
    float dy = pos[p * 3 + 1] - qy;
    float dz = pos[p * 3 + 2] - qz;
    float d2 = dx * dx + dy * dy + dz * dz;
    if (d2 <= RADIUS2 && c < KNBR) {
      nbr[q * KNBR + c] = p;
      ++c;
    }
  }
  cnt[q] = c;
  for (int t = c; t < KNBR; ++t) nbr[q * KNBR + t] = q;   // self padding
}

// ---------------------------------------------------------------------------
// Kernel 2: per-query edge MLP via WMMA + masked max-aggregate.
// Block = 64 threads = 2 waves; one query per wave.
// ---------------------------------------------------------------------------
__global__ __launch_bounds__(64) void mlp_kernel(const float* __restrict__ x,
                                                 const float* __restrict__ pos,
                                                 const int*   __restrict__ nbr,
                                                 const int*   __restrict__ cnt,
                                                 const float* __restrict__ W1,
                                                 const float* __restrict__ b1,
                                                 const float* __restrict__ W2,
                                                 const float* __restrict__ b2,
                                                 const float* __restrict__ W3,
                                                 const float* __restrict__ b3,
                                                 float* __restrict__ out) {
  // Transposed weights: sWt[n][k] so B-fragments are contiguous in k.
  __shared__ __align__(16) _Float16 sW1t[F1 * W1S];       // 13.0 KB
  __shared__ __align__(16) _Float16 sW2t[F2 * W2S];       //  9.0 KB
  __shared__ __align__(16) _Float16 sW3t[F3 * W3S];       // 18.0 KB
  __shared__ __align__(16) _Float16 sH[2][KNBR * HS];     // 18.0 KB (row-major H)
  __shared__ float sB[F1 + F2 + F3];                      //  1.0 KB
  __shared__ int   sNbr[2][KNBR];                         //  0.5 KB

  const int tid  = threadIdx.x;           // 0..63
  const int wave = tid >> 5;              // 0..1
  const int lane = tid & 31;
  const int m16  = lane & 15;
  const int half = lane >> 4;

  // ---- stage transposed weights (f32 -> f16) ------------------------------
  for (int i = tid; i < F1 * W1S; i += 64) {
    int n = i / W1S, k = i - n * W1S;
    sW1t[i] = (k < FIN + 3) ? (_Float16)W1[k * F1 + n] : (_Float16)0.0f;
  }
  for (int i = tid; i < F2 * W2S; i += 64) {
    int n = i / W2S, k = i - n * W2S;
    sW2t[i] = (k < F1) ? (_Float16)W2[k * F2 + n] : (_Float16)0.0f;
  }
  for (int i = tid; i < F3 * W3S; i += 64) {
    int n = i / W3S, k = i - n * W3S;
    sW3t[i] = (k < F2) ? (_Float16)W3[k * F3 + n] : (_Float16)0.0f;
  }
  sB[tid]            = b1[tid];
  sB[64 + tid]       = b2[tid];
  sB[128 + tid]      = b3[tid];
  sB[128 + 64 + tid] = b3[64 + tid];

  const int q = blockIdx.x * 2 + wave;    // query id
  sNbr[wave][lane]      = nbr[q * KNBR + lane];
  sNbr[wave][lane + 32] = nbr[q * KNBR + lane + 32];
  const int   count = cnt[q];
  const float qp0 = pos[q * 3 + 0];
  const float qp1 = pos[q * 3 + 1];
  const float qp2 = pos[q * 3 + 2];
  __syncthreads();

  // ======================= GEMM1: H1 = relu(A @ W1 + b1) ===================
#pragma unroll
  for (int mt = 0; mt < 4; ++mt) {
    const int row = mt * 16 + m16;
    const int j   = sNbr[wave][row];
    const v4f* xj4 = (const v4f*)(x + (size_t)j * FIN);
    v16h af[3];
#pragma unroll
    for (int kc = 0; kc < 2; ++kc) {
      int o = kc * 8 + half * 2;
      af[kc] = cvtA(xj4[o], xj4[o + 1], xj4[o + 4], xj4[o + 5]);
    }
    {   // k = 64..95 chunk: rel-pos (half 0, e<3), zeros elsewhere
      v16h a2;
#pragma unroll
      for (int e = 0; e < 16; ++e) a2[e] = (_Float16)0.0f;
      if (half == 0) {
        a2[0] = (_Float16)(pos[(size_t)j * 3 + 0] - qp0);
        a2[1] = (_Float16)(pos[(size_t)j * 3 + 1] - qp1);
        a2[2] = (_Float16)(pos[(size_t)j * 3 + 2] - qp2);
      }
      af[2] = a2;
    }
#pragma unroll
    for (int nt = 0; nt < 4; ++nt) {
      const int n = nt * 16 + m16;
      v8f acc = {0.f, 0.f, 0.f, 0.f, 0.f, 0.f, 0.f, 0.f};
#pragma unroll
      for (int kc = 0; kc < 3; ++kc) {
        v16h b = ldsB(&sW1t[n * W1S + kc * 32 + half * 8]);
        acc = __builtin_amdgcn_wmma_f32_16x16x32_f16(false, af[kc], false, b,
                                                     (short)0, acc, false, false);
      }
      float bias = sB[n];
#pragma unroll
      for (int r = 0; r < 8; ++r) {
        int   m = mt * 16 + half * 8 + r;
        float v = fmaxf(acc[r] + bias, 0.0f);
        sH[wave][m * HS + n] = (_Float16)v;
      }
    }
  }

  // ======================= GEMM2: H2 = relu(H1 @ W2 + b2) (in-place) =======
#pragma unroll
  for (int mt = 0; mt < 4; ++mt) {
    v16h af[2];
#pragma unroll
    for (int kc = 0; kc < 2; ++kc)
      af[kc] = ldsA(&sH[wave][(mt * 16 + m16) * HS + kc * 32 + half * 8]);
#pragma unroll
    for (int nt = 0; nt < 4; ++nt) {
      const int n = nt * 16 + m16;
      v8f acc = {0.f, 0.f, 0.f, 0.f, 0.f, 0.f, 0.f, 0.f};
#pragma unroll
      for (int kc = 0; kc < 2; ++kc) {
        v16h b = ldsB(&sW2t[n * W2S + kc * 32 + half * 8]);
        acc = __builtin_amdgcn_wmma_f32_16x16x32_f16(false, af[kc], false, b,
                                                     (short)0, acc, false, false);
      }
      float bias = sB[64 + n];
#pragma unroll
      for (int r = 0; r < 8; ++r) {
        int   m = mt * 16 + half * 8 + r;
        float v = fmaxf(acc[r] + bias, 0.0f);
        sH[wave][m * HS + n] = (_Float16)v;
      }
    }
  }

  // ============= GEMM3: H3 = relu(H2 @ W3 + b3), masked max over rows ======
  float pmax[8];
#pragma unroll
  for (int nt = 0; nt < 8; ++nt) pmax[nt] = -__builtin_inff();

#pragma unroll
  for (int mt = 0; mt < 4; ++mt) {
    v16h af[2];
#pragma unroll
    for (int kc = 0; kc < 2; ++kc)
      af[kc] = ldsA(&sH[wave][(mt * 16 + m16) * HS + kc * 32 + half * 8]);
#pragma unroll
    for (int nt = 0; nt < 8; ++nt) {
      const int n = nt * 16 + m16;
      v8f acc = {0.f, 0.f, 0.f, 0.f, 0.f, 0.f, 0.f, 0.f};
#pragma unroll
      for (int kc = 0; kc < 2; ++kc) {
        v16h b = ldsB(&sW3t[n * W3S + kc * 32 + half * 8]);
        acc = __builtin_amdgcn_wmma_f32_16x16x32_f16(false, af[kc], false, b,
                                                     (short)0, acc, false, false);
      }
      float bias = sB[128 + n];
#pragma unroll
      for (int r = 0; r < 8; ++r) {
        int   row = mt * 16 + half * 8 + r;
        float v   = fmaxf(acc[r] + bias, 0.0f);
        if (row < count) pmax[nt] = fmaxf(pmax[nt], v);
      }
    }
  }

  // merge the two lane-halves (column n lives in lanes n and n+16)
#pragma unroll
  for (int nt = 0; nt < 8; ++nt) {
    float other = __shfl_xor(pmax[nt], 16, 32);
    float m     = fmaxf(pmax[nt], other);
    if (half == 0) out[(size_t)q * F3 + nt * 16 + m16] = m;
  }
}

// ---------------------------------------------------------------------------
// Kernel 3: passthrough outputs (pos, batch) into the tail of d_out.
// ---------------------------------------------------------------------------
__global__ __launch_bounds__(256) void tail_kernel(const float* __restrict__ pos,
                                                   float* __restrict__ out) {
  int i = blockIdx.x * 256 + threadIdx.x;
  const int featN = NPTS * F3;
  if (i < NPTS * 3) out[featN + i] = pos[i];
  if (i < NPTS)     out[featN + NPTS * 3 + i] = (float)(i >> 12);  // batch
}

// ---------------------------------------------------------------------------
extern "C" void kernel_launch(void* const* d_in, const int* in_sizes, int n_in,
                              void* d_out, int out_size, void* d_ws, size_t ws_size,
                              hipStream_t stream) {
  const float* x   = (const float*)d_in[0];
  const float* pos = (const float*)d_in[1];
  // d_in[2] = batch (int32), recomputed on the fly
  const float* W1 = (const float*)d_in[3];
  const float* b1 = (const float*)d_in[4];
  const float* W2 = (const float*)d_in[5];
  const float* b2 = (const float*)d_in[6];
  const float* W3 = (const float*)d_in[7];
  const float* b3 = (const float*)d_in[8];
  float* out = (float*)d_out;

  int* nbr = (int*)d_ws;                       // [NPTS * KNBR]
  int* cnt = nbr + (size_t)NPTS * KNBR;        // [NPTS]

  nbr_kernel<<<NPTS / 256, 256, 0, stream>>>(pos, nbr, cnt);
  mlp_kernel<<<NPTS / 2, 64, 0, stream>>>(x, pos, nbr, cnt,
                                          W1, b1, W2, b2, W3, b3, out);
  tail_kernel<<<(NPTS * 3 + 255) / 256, 256, 0, stream>>>(pos, out);
}
